// WaveletBlock_32916629357261
// MI455X (gfx1250) — compile-verified, matrix operand order checked by reading, not run on hardware
//
#include <hip/hip_runtime.h>
#include <math.h>

// ---------------------------------------------------------------------------
// WaveletBlock for MI455X (gfx1250, wave32, WMMA).
// Memory-bound workload (~1GB traffic @ 23.3TB/s ~ 40us); ~69 of 72 GFLOP are
// GEMMs -> v_wmma_f32_16x16x32_f16 with f32 accumulation, f16 channels-last
// activations, fused bias/GELU/gate/SCA epilogues.
// ---------------------------------------------------------------------------

typedef _Float16 h8v  __attribute__((ext_vector_type(8)));
typedef _Float16 h16v __attribute__((ext_vector_type(16)));
typedef float    f8v  __attribute__((ext_vector_type(8)));

#define NB   8
#define NC   64
#define NH   256
#define NW   256
#define HH   128      // half-res H
#define HW   128      // half-res W
#define C4   256      // 4*C
#define C6   384      // 6*C
#define NPH  131072   // half-res pixels (8*128*128)
#define NPF  524288   // full-res pixels (8*256*256)

// workspace layout (bytes)
#define OFF_W1H  0u
#define OFF_W2H  196608u
#define OFF_W3H  393216u
#define OFF_W4H  401408u
#define OFF_W5H  417792u
#define OFF_PART 425984u   // pool partials: 8*32*64 f32
#define OFF_ATT  491520u   // 8*64 f32
#define OFF_A    1048576u                    // 96MB rotating region
#define OFF_G    (OFF_A + 100663296u)        // 64MB region (gate output)

// ---------------------------------------------------------------------------
// f32 -> f16 weight conversion
// ---------------------------------------------------------------------------
__global__ void cvt_f16(const float* __restrict__ s, _Float16* __restrict__ d, int n) {
    int i = blockIdx.x * 256 + threadIdx.x;
    if (i < n) d[i] = (_Float16)s[i];
}

// ---------------------------------------------------------------------------
// A/B fragment loader: 16 f16 per lane as two 16B chunks (K layout per ISA
// 7.12.2: lane-half selects K 0-7/16-23 vs 8-15/24-31; caller pre-offsets kb).
// ---------------------------------------------------------------------------
__device__ __forceinline__ h16v load_frag16(const _Float16* p) {
    h8v lo = *reinterpret_cast<const h8v*>(p);
    h8v hi = *reinterpret_cast<const h8v*>(p + 16);
    return __builtin_shufflevector(lo, hi, 0,1,2,3,4,5,6,7,8,9,10,11,12,13,14,15);
}

// ---------------------------------------------------------------------------
// WMMA GEMM core: out[m][n] = sum_k A[m][k]*Wt[n][k] + bias[n]
// block = 128 threads = 4 waves, wave computes a 16(M)x64(N) tile.
// ATT=true folds per-(batch,k) SCA scaling into the A fragment; split at
// compile time so the common path has a branch-free inner loop.
// ---------------------------------------------------------------------------
template <bool ATT>
__device__ __forceinline__ void gemm_core(
    const _Float16* __restrict__ A, const _Float16* __restrict__ Wt,
    const float* __restrict__ bias, _Float16* __restrict__ out,
    int M, int N, int K, const float* __restrict__ att)
{
    const int lane = threadIdx.x & 31;
    const int wave = threadIdx.x >> 5;
    const int m0 = (blockIdx.x * 4 + wave) * 16;
    const int n0 = blockIdx.y * 64;
    const int rowA = m0 + (lane & 15);
    const int colB = n0 + (lane & 15);
    const int khalf = (lane >> 4) * 8;

    f8v acc0 = {0,0,0,0,0,0,0,0};
    f8v acc1 = {0,0,0,0,0,0,0,0};
    f8v acc2 = {0,0,0,0,0,0,0,0};
    f8v acc3 = {0,0,0,0,0,0,0,0};

    const float* attb = ATT ? (att + (m0 >> 16) * NC) : (const float*)0;

    for (int k0 = 0; k0 < K; k0 += 32) {
        const int kb = k0 + khalf;
        h16v a = load_frag16(A + (size_t)rowA * K + kb);
        if (ATT) {
            #pragma unroll
            for (int i = 0; i < 8; ++i) {
                a[i]     = (_Float16)((float)a[i]     * attb[kb + i]);
                a[i + 8] = (_Float16)((float)a[i + 8] * attb[kb + 16 + i]);
            }
        }
        h16v bf0 = load_frag16(Wt + (size_t)(colB)      * K + kb);
        h16v bf1 = load_frag16(Wt + (size_t)(colB + 16) * K + kb);
        h16v bf2 = load_frag16(Wt + (size_t)(colB + 32) * K + kb);
        h16v bf3 = load_frag16(Wt + (size_t)(colB + 48) * K + kb);
        acc0 = __builtin_amdgcn_wmma_f32_16x16x32_f16(false, a, false, bf0, (short)0, acc0, false, false);
        acc1 = __builtin_amdgcn_wmma_f32_16x16x32_f16(false, a, false, bf1, (short)0, acc1, false, false);
        acc2 = __builtin_amdgcn_wmma_f32_16x16x32_f16(false, a, false, bf2, (short)0, acc2, false, false);
        acc3 = __builtin_amdgcn_wmma_f32_16x16x32_f16(false, a, false, bf3, (short)0, acc3, false, false);
    }

    const int rb = m0 + ((lane >> 4) << 3);   // C layout: VGPR r -> row rb+r
    const int c0 = n0 + (lane & 15);
    const float b0 = bias[c0], b1 = bias[c0 + 16], b2 = bias[c0 + 32], b3 = bias[c0 + 48];
    #pragma unroll
    for (int r = 0; r < 8; ++r) {
        size_t mr = (size_t)(rb + r) * N;
        out[mr + c0]      = (_Float16)(acc0[r] + b0);
        out[mr + c0 + 16] = (_Float16)(acc1[r] + b1);
        out[mr + c0 + 32] = (_Float16)(acc2[r] + b2);
        out[mr + c0 + 48] = (_Float16)(acc3[r] + b3);
    }
}

__global__ __launch_bounds__(128) void gemm_wmma(
    const _Float16* __restrict__ A, const _Float16* __restrict__ Wt,
    const float* __restrict__ bias, _Float16* __restrict__ out,
    int M, int N, int K)
{
    gemm_core<false>(A, Wt, bias, out, M, N, K, (const float*)0);
}

__global__ __launch_bounds__(128) void gemm_wmma_att(
    const _Float16* __restrict__ A, const _Float16* __restrict__ Wt,
    const float* __restrict__ bias, _Float16* __restrict__ out,
    int M, int N, int K, const float* __restrict__ att)
{
    gemm_core<true>(A, Wt, bias, out, M, N, K, att);
}

// ---------------------------------------------------------------------------
// Gated WMMA GEMM for w4 (N=128, K=64): wave owns all 8 N-tiles so column c
// (tiles 0..3) and c+64 (tiles 4..7) live in the same lane -> SimpleGate a*g
// fused in the epilogue; the 2C tensor never hits memory.
// ---------------------------------------------------------------------------
__global__ __launch_bounds__(128) void gemm_gate(
    const _Float16* __restrict__ A, const _Float16* __restrict__ Wt,
    const float* __restrict__ bias, _Float16* __restrict__ out)
{
    const int lane = threadIdx.x & 31;
    const int wave = threadIdx.x >> 5;
    const int m0 = (blockIdx.x * 4 + wave) * 16;
    const int rowA = m0 + (lane & 15);
    const int colB = lane & 15;
    const int khalf = (lane >> 4) * 8;

    const f8v zero = {0,0,0,0,0,0,0,0};
    f8v acc[8];
    #pragma unroll
    for (int j = 0; j < 8; ++j) acc[j] = zero;

    for (int k0 = 0; k0 < 64; k0 += 32) {
        const int kb = k0 + khalf;
        h16v a = load_frag16(A + (size_t)rowA * 64 + kb);
        #pragma unroll
        for (int j = 0; j < 8; ++j) {
            h16v bf = load_frag16(Wt + (size_t)(colB + j * 16) * 64 + kb);
            acc[j] = __builtin_amdgcn_wmma_f32_16x16x32_f16(false, a, false, bf, (short)0, acc[j], false, false);
        }
    }

    const int rb = m0 + ((lane >> 4) << 3);
    const int c0 = lane & 15;
    #pragma unroll
    for (int j = 0; j < 4; ++j) {
        const int n = j * 16 + c0;
        const float ba = bias[n], bg = bias[64 + n];
        #pragma unroll
        for (int r = 0; r < 8; ++r) {
            float av = acc[j][r] + ba;
            float gv = acc[j + 4][r] + bg;
            out[(size_t)(rb + r) * 64 + n] = (_Float16)(av * gv);
        }
    }
}

// ---------------------------------------------------------------------------
// LayerNorm2d + DWT fused. Block: (b, hh, 64-col tile) staged in 32KB LDS.
// Output: f16 channels-last [131072][256] with column = k*64+c.
// ---------------------------------------------------------------------------
__global__ __launch_bounds__(256) void ln_dwt(
    const float* __restrict__ x, const float* __restrict__ filt,
    const float* __restrict__ lw, const float* __restrict__ lb,
    _Float16* __restrict__ s)
{
    const int bid = blockIdx.x;
    const int t  = bid & 3;
    const int hh = (bid >> 2) & 127;
    const int b  = bid >> 9;
    const int col0 = t * 64;

    __shared__ float ls[64 * 2 * 64];     // [c][r][w]
    __shared__ float mu[128], rs[128];
    __shared__ float ff[16];
    if (threadIdx.x < 16) ff[threadIdx.x] = filt[threadIdx.x];

    for (int idx = threadIdx.x; idx < 64 * 2 * 64; idx += 256) {
        int c = idx >> 7, r = (idx >> 6) & 1, w = idx & 63;
        ls[idx] = x[(((size_t)b * NC + c) * NH + (hh * 2 + r)) * NW + col0 + w];
    }
    __syncthreads();

    if (threadIdx.x < 128) {
        int p = threadIdx.x;
        float su = 0.f, sq = 0.f;
        for (int c = 0; c < 64; ++c) { float v = ls[c * 128 + p]; su += v; sq += v * v; }
        float m = su * (1.f / 64.f);
        mu[p] = m;
        rs[p] = rsqrtf(sq * (1.f / 64.f) - m * m + 1e-6f);
    }
    __syncthreads();

    const int c = threadIdx.x & 63;
    const int hpb = threadIdx.x >> 6;
    const float wgt = lw[c], bia = lb[c];
    for (int i = 0; i < 8; ++i) {
        int hp = hpb + i * 4;
        float v[2][2];
        #pragma unroll
        for (int p = 0; p < 2; ++p)
            #pragma unroll
            for (int q = 0; q < 2; ++q) {
                int pi = p * 64 + hp * 2 + q;
                v[p][q] = (ls[c * 128 + pi] - mu[pi]) * rs[pi] * wgt + bia;
            }
        size_t m = ((size_t)(b * HH + hh)) * HW + (col0 >> 1) + hp;
        #pragma unroll
        for (int k = 0; k < 4; ++k) {
            float sv = ff[k*4+0]*v[0][0] + ff[k*4+1]*v[0][1] + ff[k*4+2]*v[1][0] + ff[k*4+3]*v[1][1];
            s[m * C4 + k * 64 + c] = (_Float16)sv;
        }
    }
}

// ---------------------------------------------------------------------------
// Depthwise conv (SAME, zero pad) + exact GELU. thread = channel -> every tap
// is a fully-coalesced 384-wide channels-last row.
// ---------------------------------------------------------------------------
__global__ __launch_bounds__(384) void dwconv_gelu(
    const _Float16* __restrict__ in, const float* __restrict__ wdw,
    const float* __restrict__ bias, _Float16* __restrict__ out, int R)
{
    const int c = threadIdx.x;
    const size_t pix = blockIdx.x;
    const int xw = pix & 127;
    const int yh = (pix >> 7) & 127;
    const int b  = pix >> 14;
    const int ks = 2 * R + 1;

    float acc = bias[c];
    for (int dy = -R; dy <= R; ++dy) {
        int yy = yh + dy; if (yy < 0 || yy > 127) continue;
        for (int dx = -R; dx <= R; ++dx) {
            int xx = xw + dx; if (xx < 0 || xx > 127) continue;
            float wv = wdw[(c * ks + (dy + R)) * ks + (dx + R)];
            acc += wv * (float)in[((((size_t)b * HH + yy) * HW) + xx) * C6 + c];
        }
    }
    float g = 0.5f * acc * (1.f + erff(acc * 0.70710678118654752f));
    out[pix * C6 + c] = (_Float16)g;
}

// ---------------------------------------------------------------------------
// IDWT: s2 [131072][256] f16 -> full-res channels-last f16 [524288][64]
// ---------------------------------------------------------------------------
__global__ __launch_bounds__(256) void idwt_k(
    const _Float16* __restrict__ s2, const float* __restrict__ filt,
    _Float16* __restrict__ xw)
{
    __shared__ float ff[16];
    if (threadIdx.x < 16) ff[threadIdx.x] = filt[threadIdx.x];
    __syncthreads();

    const int bid = blockIdx.x;
    const int t  = bid & 3;
    const int hh = (bid >> 2) & 127;
    const int b  = bid >> 9;
    const int c = threadIdx.x & 63;
    const int hpb = threadIdx.x >> 6;

    for (int i = 0; i < 8; ++i) {
        int hp = hpb + i * 4;
        int ww = t * 32 + hp;
        size_t m = ((size_t)(b * HH + hh)) * HW + ww;
        float sv[4];
        #pragma unroll
        for (int k = 0; k < 4; ++k) sv[k] = (float)s2[m * C4 + k * 64 + c];
        #pragma unroll
        for (int p = 0; p < 2; ++p)
            #pragma unroll
            for (int q = 0; q < 2; ++q) {
                float v = ff[0+p*2+q]*sv[0] + ff[4+p*2+q]*sv[1] + ff[8+p*2+q]*sv[2] + ff[12+p*2+q]*sv[3];
                xw[((((size_t)b * NH + hh * 2 + p) * NW) + ww * 2 + q) * NC + c] = (_Float16)v;
            }
    }
}

// ---------------------------------------------------------------------------
// SCA global-average-pool: deterministic two-stage (no float atomics).
// ---------------------------------------------------------------------------
__global__ __launch_bounds__(256) void pool_partial(
    const _Float16* __restrict__ xw, float* __restrict__ part)
{
    const int b = blockIdx.x >> 5;
    const int chunk = blockIdx.x & 31;
    const int c = threadIdx.x & 63;
    const int g = threadIdx.x >> 6;
    size_t base = (size_t)b * 65536 + chunk * 2048;
    float s = 0.f;
    for (int j = 0; j < 512; ++j)
        s += (float)xw[(base + g + j * 4) * NC + c];
    __shared__ float ls[256];
    ls[threadIdx.x] = s;
    __syncthreads();
    if (threadIdx.x < 64)
        part[(size_t)blockIdx.x * 64 + threadIdx.x] =
            ls[threadIdx.x] + ls[64 + threadIdx.x] + ls[128 + threadIdx.x] + ls[192 + threadIdx.x];
}

__global__ __launch_bounds__(64) void sca_att(
    const float* __restrict__ part, const float* __restrict__ wsca,
    const float* __restrict__ bsca, float* __restrict__ att)
{
    const int b = blockIdx.x;
    const int o = threadIdx.x;
    __shared__ float pool[64];
    float s = 0.f;
    for (int ch = 0; ch < 32; ++ch) s += part[((size_t)b * 32 + ch) * 64 + o];
    pool[o] = s * (1.f / 65536.f);
    __syncthreads();
    float a = bsca[o];
    for (int cc = 0; cc < 64; ++cc) a += wsca[o * 64 + cc] * pool[cc];
    att[b * 64 + o] = a;
}

// ---------------------------------------------------------------------------
// y = x_in + x3*beta, written NCHW straight into d_out (LDS transpose of x3).
// ---------------------------------------------------------------------------
__global__ __launch_bounds__(256) void residual1(
    const float* __restrict__ xin, const _Float16* __restrict__ x3,
    const float* __restrict__ beta, float* __restrict__ yout)
{
    const int bid = blockIdx.x;
    const int t  = bid & 3;
    const int hc = (bid >> 2) & 255;
    const int b  = bid >> 10;
    const int w0 = t * 64;
    __shared__ float ls[64 * 65];
    size_t pixbase = ((size_t)b * NH + hc) * NW + w0;

    for (int idx = threadIdx.x; idx < 4096; idx += 256) {
        int c = idx & 63, w = idx >> 6;
        ls[c * 65 + w] = (float)x3[(pixbase + w) * NC + c];
    }
    __syncthreads();
    for (int idx = threadIdx.x; idx < 4096; idx += 256) {
        int w = idx & 63, c = idx >> 6;
        size_t o = (((size_t)b * NC + c) * NH + hc) * NW + w0 + w;
        yout[o] = xin[o] + ls[c * 65 + w] * beta[c];
    }
}

// ---------------------------------------------------------------------------
// LayerNorm2d over y (NCHW in d_out) -> f16 channels-last xn.
// ---------------------------------------------------------------------------
__global__ __launch_bounds__(256) void ln2_k(
    const float* __restrict__ y, const float* __restrict__ lw,
    const float* __restrict__ lb, _Float16* __restrict__ xn)
{
    const int bid = blockIdx.x;
    const int t  = bid & 3;
    const int hc = (bid >> 2) & 255;
    const int b  = bid >> 10;
    const int w0 = t * 64;
    __shared__ float ls[64 * 65];
    __shared__ float mu[64], rs[64];

    for (int idx = threadIdx.x; idx < 4096; idx += 256) {
        int w = idx & 63, c = idx >> 6;
        ls[w * 65 + c] = y[(((size_t)b * NC + c) * NH + hc) * NW + w0 + w];
    }
    __syncthreads();
    if (threadIdx.x < 64) {
        int w = threadIdx.x;
        float su = 0.f, sq = 0.f;
        for (int c = 0; c < 64; ++c) { float v = ls[w * 65 + c]; su += v; sq += v * v; }
        float m = su * (1.f / 64.f);
        mu[w] = m;
        rs[w] = rsqrtf(sq * (1.f / 64.f) - m * m + 1e-6f);
    }
    __syncthreads();
    size_t pixbase = ((size_t)b * NH + hc) * NW + w0;
    for (int idx = threadIdx.x; idx < 4096; idx += 256) {
        int c = idx & 63, w = idx >> 6;
        xn[(pixbase + w) * NC + c] =
            (_Float16)((ls[w * 65 + c] - mu[w]) * rs[w] * lw[c] + lb[c]);
    }
}

// ---------------------------------------------------------------------------
// out = y + z*gamma, in-place on d_out (LDS transpose of z).
// ---------------------------------------------------------------------------
__global__ __launch_bounds__(256) void residual2(
    const _Float16* __restrict__ z, const float* __restrict__ gamma,
    float* __restrict__ out)
{
    const int bid = blockIdx.x;
    const int t  = bid & 3;
    const int hc = (bid >> 2) & 255;
    const int b  = bid >> 10;
    const int w0 = t * 64;
    __shared__ float ls[64 * 65];
    size_t pixbase = ((size_t)b * NH + hc) * NW + w0;

    for (int idx = threadIdx.x; idx < 4096; idx += 256) {
        int c = idx & 63, w = idx >> 6;
        ls[c * 65 + w] = (float)z[(pixbase + w) * NC + c];
    }
    __syncthreads();
    for (int idx = threadIdx.x; idx < 4096; idx += 256) {
        int w = idx & 63, c = idx >> 6;
        size_t o = (((size_t)b * NC + c) * NH + hc) * NW + w0 + w;
        out[o] = out[o] + ls[c * 65 + w] * gamma[c];
    }
}

// ---------------------------------------------------------------------------
extern "C" void kernel_launch(void* const* d_in, const int* in_sizes, int n_in,
                              void* d_out, int out_size, void* d_ws, size_t ws_size,
                              hipStream_t stream) {
    const float* x_in = (const float*)d_in[0];
    const float* filt = (const float*)d_in[1];
    const float* ln1w = (const float*)d_in[2];
    const float* ln1b = (const float*)d_in[3];
    const float* w1   = (const float*)d_in[4];
    const float* b1   = (const float*)d_in[5];
    const float* dw5  = (const float*)d_in[6];
    const float* bdw5 = (const float*)d_in[7];
    const float* dw3  = (const float*)d_in[8];
    const float* bdw3 = (const float*)d_in[9];
    const float* w2   = (const float*)d_in[10];
    const float* b2   = (const float*)d_in[11];
    const float* wsca = (const float*)d_in[12];
    const float* bsca = (const float*)d_in[13];
    const float* w3   = (const float*)d_in[14];
    const float* b3   = (const float*)d_in[15];
    const float* beta = (const float*)d_in[16];
    const float* ln2w = (const float*)d_in[17];
    const float* ln2b = (const float*)d_in[18];
    const float* w4   = (const float*)d_in[19];
    const float* b4   = (const float*)d_in[20];
    const float* w5   = (const float*)d_in[21];
    const float* b5   = (const float*)d_in[22];
    const float* gamma= (const float*)d_in[23];
    (void)in_sizes; (void)n_in; (void)out_size; (void)ws_size;

    char* ws = (char*)d_ws;
    _Float16* w1h  = (_Float16*)(ws + OFF_W1H);
    _Float16* w2h  = (_Float16*)(ws + OFF_W2H);
    _Float16* w3h  = (_Float16*)(ws + OFF_W3H);
    _Float16* w4h  = (_Float16*)(ws + OFF_W4H);
    _Float16* w5h  = (_Float16*)(ws + OFF_W5H);
    float*    part = (float*)(ws + OFF_PART);
    float*    attp = (float*)(ws + OFF_ATT);
    _Float16* regA = (_Float16*)(ws + OFF_A);   // 96MB rotating scratch
    _Float16* regG = (_Float16*)(ws + OFF_G);   // 64MB (gate output)
    float*    outF = (float*)d_out;             // final output; also f16 scratch early
    _Float16* outH = (_Float16*)d_out;

    // 0) weights -> f16
    cvt_f16<<<(C6*C4 + 255)/256, 256, 0, stream>>>(w1, w1h, C6*C4);
    cvt_f16<<<(C4*C6 + 255)/256, 256, 0, stream>>>(w2, w2h, C4*C6);
    cvt_f16<<<(NC*NC + 255)/256, 256, 0, stream>>>(w3, w3h, NC*NC);
    cvt_f16<<<(128*NC + 255)/256, 256, 0, stream>>>(w4, w4h, 128*NC);
    cvt_f16<<<(NC*NC + 255)/256, 256, 0, stream>>>(w5, w5h, NC*NC);

    // 1) LN1 + DWT -> s_mat (regA, f16 [131072][256])
    ln_dwt<<<NB*128*4, 256, 0, stream>>>(x_in, filt, ln1w, ln1b, regA);

    // 2) pw1: s_mat x w1 -> t (d_out scratch, f16 [131072][384])
    gemm_wmma<<<dim3(NPH/64, C6/64), 128, 0, stream>>>(regA, w1h, b1, outH,
                                                       NPH, C6, C4);
    // 3) dw5 + GELU -> u (regA)
    dwconv_gelu<<<NPH, C6, 0, stream>>>(outH, dw5, bdw5, regA, 2);
    // 4) dw3 + GELU -> v (d_out scratch)
    dwconv_gelu<<<NPH, C6, 0, stream>>>(regA, dw3, bdw3, outH, 1);
    // 5) pw2: v x w2 -> s2 (regA, f16 [131072][256])
    gemm_wmma<<<dim3(NPH/64, C4/64), 128, 0, stream>>>(outH, w2h, b2, regA,
                                                       NPH, C4, C6);
    // 6) IDWT -> xw (d_out scratch, f16 [524288][64])
    idwt_k<<<NB*128*4, 256, 0, stream>>>(regA, filt, outH);

    // 7) SCA: pool + att
    pool_partial<<<NB*32, 256, 0, stream>>>(outH, part);
    sca_att<<<NB, 64, 0, stream>>>(part, wsca, bsca, attp);

    // 8) pw3 with fused SCA scaling: (xw*att) x w3 -> x3 (regA)
    gemm_wmma_att<<<dim3(NPF/64, 1), 128, 0, stream>>>(outH, w3h, b3, regA,
                                                       NPF, NC, NC, attp);
    // 9) y = x_in + x3*beta -> d_out (NCHW f32)
    residual1<<<NB*NH*4, 256, 0, stream>>>(x_in, regA, beta, outF);
    // 10) LN2 -> xn (regA, f16 [524288][64])
    ln2_k<<<NB*NH*4, 256, 0, stream>>>(outF, ln2w, ln2b, regA);
    // 11) pw4 + SimpleGate fused -> gmat (regG, f16 [524288][64])
    gemm_gate<<<NPF/64, 128, 0, stream>>>(regA, w4h, b4, regG);
    // 12) pw5: gmat x w5 -> z (regA)
    gemm_wmma<<<dim3(NPF/64, 1), 128, 0, stream>>>(regG, w5h, b5, regA,
                                                   NPF, NC, NC);
    // 13) out = y + z*gamma (in-place on d_out)
    residual2<<<NB*NH*4, 256, 0, stream>>>(regA, gamma, outF);
}
